// GATE_GCN_29411936043365
// MI455X (gfx1250) — matrix operand order repaired and checked by reading, not compiled, not run on hardware
//
#include <hip/hip_runtime.h>
#include <math.h>

#define NN 100000
#define NE 1600000
#define DIN 128
#define DH 64

typedef __attribute__((ext_vector_type(16))) __bf16 v16bf;
typedef __attribute__((ext_vector_type(8)))  float  v8f;

union BF16x16 { v16bf v; uint4 q[2]; };

__device__ __forceinline__ float elu(float v) {
    return v > 0.0f ? v : expm1f(v);
}

// ---------------- utility kernels ----------------
__global__ void zero_f32(float* __restrict__ p, long long n) {
    long long i = (long long)blockIdx.x * blockDim.x + threadIdx.x;
    if (i < n) p[i] = 0.0f;
}

__global__ void degree_kernel(const int* __restrict__ src, const int* __restrict__ dst,
                              float* __restrict__ deg_out, float* __restrict__ deg_in) {
    int e = blockIdx.x * blockDim.x + threadIdx.x;
    if (e < NE) {
        atomicAdd(&deg_out[src[e]], 1.0f);
        atomicAdd(&deg_in[dst[e]], 1.0f);
    }
}

__global__ void invsqrt_kernel(float* __restrict__ a, float* __restrict__ b, int n) {
    int i = blockIdx.x * blockDim.x + threadIdx.x;
    if (i < n) {
        a[i] = rsqrtf(fmaxf(a[i], 1.0f));
        b[i] = rsqrtf(fmaxf(b[i], 1.0f));
    }
}

// W: [K][Nout] f32 row-major  ->  Wt: [Nout][K] bf16 (transposed, WMMA-B friendly)
__global__ void prep_weight(const float* __restrict__ W, __bf16* __restrict__ Wt,
                            int K, int Nout) {
    int i = blockIdx.x * blockDim.x + threadIdx.x;
    if (i < K * Nout) {
        int k = i / Nout, n = i % Nout;
        Wt[n * K + k] = (__bf16)W[i];
    }
}

// ---------------- WMMA GEMM: Y[n,0:64] = f(X[n,:]) @ W  (+bias) ----------------
// IN_MODE: 0 = identity, 1 = scale by rowscale[n], 2 = ELU(input)
// Block = 128 threads = 4 waves; each wave computes a 16x64 tile (4 x v_wmma per K-step).
template<int KDIM, int IN_MODE, bool ADD_BIAS>
__global__ __launch_bounds__(128) void gemm_wmma(
    const float* __restrict__ X, const float* __restrict__ rowscale,
    const __bf16* __restrict__ Wt, const float* __restrict__ bias,
    float* __restrict__ Y, int nrows)
{
    const int lane = threadIdx.x & 31;
    const int wave = threadIdx.x >> 5;
    const int half = lane >> 4;       // lane group 0 / 1
    const int l16  = lane & 15;
    const int rowBase = blockIdx.x * 64 + wave * 16;
    // Clamp instead of predicating the loads: OOB lanes fetch a valid row whose
    // WMMA result lands only in D rows we never store (row M depends only on A row M).
    const int arow = (rowBase + l16 < nrows) ? (rowBase + l16) : (nrows - 1);

    float scale = 1.0f;
    if (IN_MODE == 1) scale = rowscale[arow];

    const float* __restrict__ xrow = X + (long long)arow * KDIM;

    v8f acc[4] = {};   // 4 column tiles of 16x16 f32

    #pragma unroll
    for (int kb = 0; kb < KDIM; kb += 32) {
        // A fragment (16-bit A 16x32 layout): lane = row M=l16; elements map to two
        // contiguous 8-float runs: K in [kb+half*8, +8) and [kb+16+half*8, +8).
        const float4* r0 = (const float4*)(xrow + kb + half * 8);
        const float4* r1 = (const float4*)(xrow + kb + 16 + half * 8);
        float a[16];
        { float4 t0 = r0[0], t1 = r0[1], t2 = r1[0], t3 = r1[1];
          a[0]=t0.x; a[1]=t0.y; a[2]=t0.z; a[3]=t0.w;
          a[4]=t1.x; a[5]=t1.y; a[6]=t1.z; a[7]=t1.w;
          a[8]=t2.x; a[9]=t2.y; a[10]=t2.z; a[11]=t2.w;
          a[12]=t3.x; a[13]=t3.y; a[14]=t3.z; a[15]=t3.w; }
        BF16x16 A;
        #pragma unroll
        for (int e = 0; e < 16; ++e) {
            float v = a[e];
            if (IN_MODE == 1) v *= scale;
            else if (IN_MODE == 2) v = elu(v);
            A.v[e] = (__bf16)v;            // backend packs via v_cvt_pk_bf16_f32
        }
        // B fragments: lane = column N=l16 of tile c; K runs [kb+half*8, +8) and [kb+16+half*8, +8) (matches A)
        #pragma unroll
        for (int c = 0; c < 4; ++c) {
            const int col = c * 16 + l16;
            BF16x16 B;
            const uint4* bp = (const uint4*)(Wt + (long long)col * KDIM + kb + half * 8);
            B.q[0] = bp[0];
            B.q[1] = bp[2];
            acc[c] = __builtin_amdgcn_wmma_f32_16x16x32_bf16(
                         false, A.v, false, B.v, (short)0, acc[c], false, false);
        }
    }

    // Store D: element r -> row = rowBase + half*8 + r, col = c*16 + l16
    #pragma unroll
    for (int c = 0; c < 4; ++c) {
        const int col = c * 16 + l16;
        const float bv = ADD_BIAS ? bias[col] : 0.0f;
        #pragma unroll
        for (int r = 0; r < 8; ++r) {
            const int row = rowBase + half * 8 + r;
            if (row < nrows) Y[(long long)row * DH + col] = acc[c][r] + bv;
        }
    }
}

// ---------------- SpMM: AGG[dst] += H[src], wave-per-edge, float2 per lane ----------------
__global__ __launch_bounds__(256) void spmm_kernel(
    const int* __restrict__ src, const int* __restrict__ dst,
    const float* __restrict__ H, float* __restrict__ AGG)
{
    const long long t = (long long)blockIdx.x * blockDim.x + threadIdx.x;
    const long long e = t >> 5;
    const int lane = (int)(t & 31);
    if (e < NE) {
        const int s = src[e], d = dst[e];
        const float2 v = *(const float2*)(H + (long long)s * DH + 2 * lane);
        float* ad = AGG + (long long)d * DH + 2 * lane;
        atomicAdd(&ad[0], v.x);
        atomicAdd(&ad[1], v.y);
    }
}

// ---------------- OUT[n,f] = act(AGG[n,f]*inv_in[n] + bias[f]) ----------------
template<bool ELU_ON>
__global__ void finish_kernel(const float* __restrict__ AGG, const float* __restrict__ inv_in,
                              const float* __restrict__ bias, float* __restrict__ OUT)
{
    const long long i = (long long)blockIdx.x * blockDim.x + threadIdx.x;
    if (i < (long long)NN * DH) {
        const int n = (int)(i >> 6);
        const int f = (int)(i & 63);
        float v = AGG[i] * inv_in[n] + bias[f];
        if (ELU_ON) v = elu(v);
        OUT[i] = v;
    }
}

extern "C" void kernel_launch(void* const* d_in, const int* in_sizes, int n_in,
                              void* d_out, int out_size, void* d_ws, size_t ws_size,
                              hipStream_t stream) {
    (void)in_sizes; (void)n_in; (void)out_size; (void)ws_size;
    const float* x   = (const float*)d_in[0];
    const int*   src = (const int*)  d_in[1];
    const int*   dst = (const int*)  d_in[2];
    const float* W0  = (const float*)d_in[3];
    const float* b0  = (const float*)d_in[4];
    const float* W1  = (const float*)d_in[5];
    const float* b1  = (const float*)d_in[6];
    const float* W2  = (const float*)d_in[7];
    const float* b2  = (const float*)d_in[8];
    const float* Wl  = (const float*)d_in[9];
    const float* bl  = (const float*)d_in[10];

    // workspace layout (256B-aligned regions)
    char* ws = (char*)d_ws;
    size_t off = 0;
    auto alloc = [&](size_t bytes) { size_t o = off; off += (bytes + 255) & ~(size_t)255; return o; };
    float*  inv_out = (float*)(ws + alloc((size_t)NN * 4));
    float*  inv_in  = (float*)(ws + alloc((size_t)NN * 4));
    float*  agg     = (float*)(ws + alloc((size_t)NN * DH * 4));
    float*  hbuf    = (float*)(ws + alloc((size_t)NN * DH * 4));
    __bf16* W0t     = (__bf16*)(ws + alloc((size_t)DIN * DH * 2));
    __bf16* W1t     = (__bf16*)(ws + alloc((size_t)DH  * DH * 2));
    __bf16* W2t     = (__bf16*)(ws + alloc((size_t)DH  * DH * 2));
    __bf16* Wlt     = (__bf16*)(ws + alloc((size_t)DH  * DH * 2));
    float*  proj    = (float*)d_out;   // projection scratch; overwritten by final GEMM

    const long long nf = (long long)NN * DH;
    const int zb  = (int)((nf + 255) / 256);
    const int nb  = (NN + 255) / 256;
    const int eb  = (NE + 255) / 256;
    const int spb = (int)(((long long)NE * 32 + 255) / 256);
    const int gb  = (NN + 63) / 64;

    // degrees -> inv sqrt
    zero_f32<<<nb, 256, 0, stream>>>(inv_out, NN);
    zero_f32<<<nb, 256, 0, stream>>>(inv_in,  NN);
    degree_kernel<<<eb, 256, 0, stream>>>(src, dst, inv_out, inv_in);
    invsqrt_kernel<<<nb, 256, 0, stream>>>(inv_out, inv_in, NN);

    // weight prep (f32 -> transposed bf16)
    prep_weight<<<(DIN * DH + 255) / 256, 256, 0, stream>>>(W0, W0t, DIN, DH);
    prep_weight<<<(DH * DH + 255) / 256,  256, 0, stream>>>(W1, W1t, DH, DH);
    prep_weight<<<(DH * DH + 255) / 256,  256, 0, stream>>>(W2, W2t, DH, DH);
    prep_weight<<<(DH * DH + 255) / 256,  256, 0, stream>>>(Wl, Wlt, DH, DH);

    // layer 0: (x * inv_out) @ W0 -> spmm -> *inv_in + b0 -> ELU
    gemm_wmma<DIN, 1, false><<<gb, 128, 0, stream>>>(x, inv_out, W0t, nullptr, proj, NN);
    zero_f32<<<zb, 256, 0, stream>>>(agg, nf);
    spmm_kernel<<<spb, 256, 0, stream>>>(src, dst, proj, agg);
    finish_kernel<true><<<zb, 256, 0, stream>>>(agg, inv_in, b0, hbuf);

    // layer 1
    gemm_wmma<DH, 1, false><<<gb, 128, 0, stream>>>(hbuf, inv_out, W1t, nullptr, proj, NN);
    zero_f32<<<zb, 256, 0, stream>>>(agg, nf);
    spmm_kernel<<<spb, 256, 0, stream>>>(src, dst, proj, agg);
    finish_kernel<true><<<zb, 256, 0, stream>>>(agg, inv_in, b1, hbuf);

    // layer 2 (no ELU inside the conv)
    gemm_wmma<DH, 1, false><<<gb, 128, 0, stream>>>(hbuf, inv_out, W2t, nullptr, proj, NN);
    zero_f32<<<zb, 256, 0, stream>>>(agg, nf);
    spmm_kernel<<<spb, 256, 0, stream>>>(src, dst, proj, agg);
    finish_kernel<false><<<zb, 256, 0, stream>>>(agg, inv_in, b2, hbuf);

    // final: elu(h) @ Wl + bl -> d_out
    gemm_wmma<DH, 2, true><<<gb, 128, 0, stream>>>(hbuf, nullptr, Wlt, bl, (float*)d_out, NN);
}
